// APPNP_29368986370487
// MI455X (gfx1250) — matrix operand order repaired and checked by reading, not compiled, not run on hardware
//
#include <hip/hip_runtime.h>
#include <hip/hip_bf16.h>

#define N_FEAT 512
#define N_HID  256
#define N_CLS  64
#define N_ITERS 10
#define ALPHA_F 0.1f
#define BETA_F  0.9f

typedef __attribute__((ext_vector_type(16))) __bf16 v16bf;
typedef __attribute__((ext_vector_type(8)))  __bf16 v8bf;
typedef __attribute__((ext_vector_type(8)))  float  v8f;

// round-to-nearest-even f32 -> bf16 (stored as u16)
__device__ __forceinline__ unsigned short f2bf(float f) {
    unsigned u = __builtin_bit_cast(unsigned, f);
    u += 0x7FFFu + ((u >> 16) & 1u);
    return (unsigned short)(u >> 16);
}

// ---------------------------------------------------------------------------
// Weight pre-transpose + bf16 conversion: W1[512,256] -> W1t[256][512] bf16
// ---------------------------------------------------------------------------
__global__ void conv_w1_kernel(const float* __restrict__ W1,
                               unsigned short* __restrict__ W1t) {
    int idx = blockIdx.x * blockDim.x + threadIdx.x;
    if (idx >= N_FEAT * N_HID) return;
    int k = idx / N_HID, n = idx % N_HID;
    W1t[(size_t)n * N_FEAT + k] = f2bf(W1[idx]);
}

__global__ void conv_w2_kernel(const float* __restrict__ W2,
                               unsigned short* __restrict__ W2t) {
    int idx = blockIdx.x * blockDim.x + threadIdx.x;
    if (idx >= N_HID * N_CLS) return;
    int k = idx / N_CLS, n = idx % N_CLS;
    W2t[(size_t)n * N_HID + k] = f2bf(W2[idx]);
}

// ---------------------------------------------------------------------------
// Fused MLP: z = relu(x@W1 + b1) @ W2 + b2, one 16-row tile per block.
// 256 threads = 8 waves. Uses v_wmma_f32_16x16x32_bf16.
// Requires gridDim.x*16 == N (N=100000 is a multiple of 16).
// ---------------------------------------------------------------------------
#define SA_STRIDE 520   // 512 + 8 bf16 pad (byte stride 1040, 16B aligned)
#define SH_STRIDE 264   // 256 + 8 bf16 pad (byte stride 528, 16B aligned)

__global__ __launch_bounds__(256)
void fused_mlp_wmma_kernel(const float* __restrict__ x,
                           const unsigned short* __restrict__ W1t,
                           const float* __restrict__ b1,
                           const unsigned short* __restrict__ W2t,
                           const float* __restrict__ b2,
                           float* __restrict__ z) {
    __shared__ unsigned short sA[16 * SA_STRIDE];  // x tile, bf16
    __shared__ unsigned short sH[16 * SH_STRIDE];  // hidden tile, bf16

    const int tid  = threadIdx.x;
    const int lane = tid & 31;
    const int wave = tid >> 5;
    const int half = lane >> 4;   // lanes 16..31 hold K upper halves
    const int lrow = lane & 15;   // A row / B-D column within tile
    const int row0 = blockIdx.x * 16;

    // ---- stage 16x512 x tile into LDS as bf16 (row r, 32-col segment) ----
    {
        int r = tid >> 4, seg = tid & 15;
        const float* src = x + (size_t)(row0 + r) * N_FEAT + seg * 32;
        unsigned short* dst = &sA[r * SA_STRIDE + seg * 32];
        #pragma unroll
        for (int i = 0; i < 32; i += 4) {
            float4 v = *(const float4*)(src + i);
            dst[i + 0] = f2bf(v.x);
            dst[i + 1] = f2bf(v.y);
            dst[i + 2] = f2bf(v.z);
            dst[i + 3] = f2bf(v.w);
        }
    }
    __syncthreads();

    // ---- GEMM1: each wave owns two 16x16 N-tiles of the 16x256 h tile ----
    const int n0a = wave * 32;
    const int n0b = n0a + 16;
    v8f acc0 = {};
    v8f acc1 = {};
    for (int k0 = 0; k0 < N_FEAT; k0 += 32) {
        // A fragment (16x32 bf16 layout): lane holds row lrow,
        // K = k0+half*16 .. +15
        const unsigned short* ap = &sA[lrow * SA_STRIDE + k0 + half * 16];
        v8bf a0 = *(const v8bf*)ap;
        v8bf a1 = *(const v8bf*)(ap + 8);
        v16bf a = __builtin_shufflevector(a0, a1,
                    0,1,2,3,4,5,6,7,8,9,10,11,12,13,14,15);
        // B fragment (32x16): lane holds column lrow, K = k0+half*16 .. +15
        v16bf bA = *(const v16bf*)&W1t[(size_t)(n0a + lrow) * N_FEAT + k0 + half * 16];
        v16bf bB = *(const v16bf*)&W1t[(size_t)(n0b + lrow) * N_FEAT + k0 + half * 16];
        acc0 = __builtin_amdgcn_wmma_f32_16x16x32_bf16(
                   false, a, false, bA, (short)0, acc0, false, false);
        acc1 = __builtin_amdgcn_wmma_f32_16x16x32_bf16(
                   false, a, false, bB, (short)0, acc1, false, false);
    }

    // bias + relu, restage hidden tile to LDS as bf16
    {
        float biasA = b1[n0a + lrow];
        float biasB = b1[n0b + lrow];
        #pragma unroll
        for (int j = 0; j < 8; ++j) {
            int r = half * 8 + j;   // D layout: VGPR j -> row half*8+j, col lrow
            float va = fmaxf(acc0[j] + biasA, 0.0f);
            float vb = fmaxf(acc1[j] + biasB, 0.0f);
            sH[r * SH_STRIDE + n0a + lrow] = f2bf(va);
            sH[r * SH_STRIDE + n0b + lrow] = f2bf(vb);
        }
    }
    __syncthreads();

    // ---- GEMM2: waves 0..3 produce the 16x64 z tile ----
    if (wave < 4) {
        const int n0 = wave * 16;
        v8f acc = {};
        for (int k0 = 0; k0 < N_HID; k0 += 32) {
            const unsigned short* ap = &sH[lrow * SH_STRIDE + k0 + half * 16];
            v8bf a0 = *(const v8bf*)ap;
            v8bf a1 = *(const v8bf*)(ap + 8);
            v16bf a = __builtin_shufflevector(a0, a1,
                        0,1,2,3,4,5,6,7,8,9,10,11,12,13,14,15);
            v16bf b = *(const v16bf*)&W2t[(size_t)(n0 + lrow) * N_HID + k0 + half * 16];
            acc = __builtin_amdgcn_wmma_f32_16x16x32_bf16(
                      false, a, false, b, (short)0, acc, false, false);
        }
        float bias = b2[n0 + lrow];
        #pragma unroll
        for (int j = 0; j < 8; ++j) {
            int r = row0 + half * 8 + j;
            z[(size_t)r * N_CLS + n0 + lrow] = acc[j] + bias;
        }
    }
}

// ---------------------------------------------------------------------------
// Degree / normalization
// ---------------------------------------------------------------------------
__global__ void init_deg_kernel(float* __restrict__ deg, int n) {
    int i = blockIdx.x * blockDim.x + threadIdx.x;
    if (i < n) deg[i] = 1.0f;   // self-loop contributes 1 to every node
}

__global__ void deg_scatter_kernel(float* __restrict__ deg,
                                   const int* __restrict__ dst, int E) {
    int e = blockIdx.x * blockDim.x + threadIdx.x;
    if (e < E) atomicAdd(&deg[dst[e]], 1.0f);
}

__global__ void dinv_kernel(float* __restrict__ deg_dinv, int n) {
    int i = blockIdx.x * blockDim.x + threadIdx.x;
    if (i < n) deg_dinv[i] = rsqrtf(deg_dinv[i]);   // deg >= 1 always
}

__global__ void edge_norm_kernel(float* __restrict__ norm,
                                 const int* __restrict__ src,
                                 const int* __restrict__ dst,
                                 const float* __restrict__ dinv, int E) {
    int e = blockIdx.x * blockDim.x + threadIdx.x;
    if (e < E) norm[e] = dinv[src[e]] * dinv[dst[e]];
}

// ---------------------------------------------------------------------------
// Propagation step: out = alpha*z + beta*(dinv^2 * h  +  scatter(norm*h[src]))
// init kernel fully overwrites out (teleport + self-loop term), then edge
// scatter accumulates with f32 global atomics (L2-resident working set).
// ---------------------------------------------------------------------------
__global__ __launch_bounds__(256)
void prop_init_kernel(float* __restrict__ out,
                      const float* __restrict__ z,
                      const float* __restrict__ hin,
                      const float* __restrict__ dinv, int n) {
    int idx = blockIdx.x * blockDim.x + threadIdx.x;   // over n*16 float4s
    if (idx >= n * 16) return;
    int node = idx >> 4;
    int c = (idx & 15) * 4;
    float di = dinv[node];
    float w = BETA_F * di * di;
    float4 zv = *(const float4*)(z   + (size_t)node * N_CLS + c);
    float4 hv = *(const float4*)(hin + (size_t)node * N_CLS + c);
    float4 o;
    o.x = ALPHA_F * zv.x + w * hv.x;
    o.y = ALPHA_F * zv.y + w * hv.y;
    o.z = ALPHA_F * zv.z + w * hv.z;
    o.w = ALPHA_F * zv.w + w * hv.w;
    *(float4*)(out + (size_t)node * N_CLS + c) = o;
}

__global__ __launch_bounds__(256)
void prop_scatter_kernel(float* __restrict__ out,
                         const float* __restrict__ hin,
                         const int* __restrict__ src,
                         const int* __restrict__ dst,
                         const float* __restrict__ norm, int E) {
    int t = blockIdx.x * blockDim.x + threadIdx.x;
    int e = t >> 4;
    int c = (t & 15) * 4;
    if (e >= E) return;
    int s = src[e], d = dst[e];
    float w = BETA_F * norm[e];
    float4 hv = *(const float4*)(hin + (size_t)s * N_CLS + c);
    float* o = out + (size_t)d * N_CLS + c;
    atomicAdd(o + 0, w * hv.x);
    atomicAdd(o + 1, w * hv.y);
    atomicAdd(o + 2, w * hv.z);
    atomicAdd(o + 3, w * hv.w);
}

// ---------------------------------------------------------------------------
// Row-wise log-softmax over 64 classes; one wave32 per row.
// ---------------------------------------------------------------------------
__global__ __launch_bounds__(256)
void log_softmax_kernel(float* __restrict__ out,
                        const float* __restrict__ in, int n) {
    int row  = blockIdx.x * 8 + (threadIdx.x >> 5);
    int lane = threadIdx.x & 31;
    if (row >= n) return;
    const float* p = in + (size_t)row * N_CLS;
    float a = p[lane];
    float b = p[lane + 32];
    float mx = fmaxf(a, b);
    #pragma unroll
    for (int m = 16; m > 0; m >>= 1) mx = fmaxf(mx, __shfl_xor(mx, m, 32));
    float s = __expf(a - mx) + __expf(b - mx);
    #pragma unroll
    for (int m = 16; m > 0; m >>= 1) s += __shfl_xor(s, m, 32);
    float lse = mx + __logf(s);
    out[(size_t)row * N_CLS + lane]      = a - lse;
    out[(size_t)row * N_CLS + lane + 32] = b - lse;
}

// ---------------------------------------------------------------------------
// Host launcher
// ---------------------------------------------------------------------------
extern "C" void kernel_launch(void* const* d_in, const int* in_sizes, int n_in,
                              void* d_out, int out_size, void* d_ws, size_t ws_size,
                              hipStream_t stream) {
    const float* x  = (const float*)d_in[0];
    const int*   ei = (const int*)d_in[1];
    const float* W1 = (const float*)d_in[2];
    const float* b1 = (const float*)d_in[3];
    const float* W2 = (const float*)d_in[4];
    const float* b2 = (const float*)d_in[5];
    float* out = (float*)d_out;

    const int N = in_sizes[0] / N_FEAT;   // 100000 (multiple of 16)
    const int E = in_sizes[1] / 2;        // 3200000
    const int* srcI = ei;
    const int* dstI = ei + E;

    // carve workspace (hipMalloc base is 256B aligned)
    char* wp = (char*)d_ws;
    auto carve = [&](size_t bytes) -> char* {
        char* r = wp;
        wp += (bytes + 255) & ~(size_t)255;
        return r;
    };
    float*          z    = (float*)carve((size_t)N * N_CLS * sizeof(float));
    float*          pA   = (float*)carve((size_t)N * N_CLS * sizeof(float));
    float*          pB   = (float*)carve((size_t)N * N_CLS * sizeof(float));
    float*          dinv = (float*)carve((size_t)N * sizeof(float));
    float*          norm = (float*)carve((size_t)E * sizeof(float));
    unsigned short* W1t  = (unsigned short*)carve((size_t)N_FEAT * N_HID * 2);
    unsigned short* W2t  = (unsigned short*)carve((size_t)N_HID * N_CLS * 2);

    // 1) weights -> transposed bf16
    conv_w1_kernel<<<(N_FEAT * N_HID + 255) / 256, 256, 0, stream>>>(W1, W1t);
    conv_w2_kernel<<<(N_HID * N_CLS + 255) / 256, 256, 0, stream>>>(W2, W2t);

    // 2) fused WMMA MLP -> z
    fused_mlp_wmma_kernel<<<N / 16, 256, 0, stream>>>(x, W1t, b1, W2t, b2, z);

    // 3) degrees + symmetric normalization
    init_deg_kernel<<<(N + 255) / 256, 256, 0, stream>>>(dinv, N);
    deg_scatter_kernel<<<(E + 255) / 256, 256, 0, stream>>>(dinv, dstI, E);
    dinv_kernel<<<(N + 255) / 256, 256, 0, stream>>>(dinv, N);
    edge_norm_kernel<<<(E + 255) / 256, 256, 0, stream>>>(norm, srcI, dstI, dinv, E);

    // 4) 10 APPNP iterations (ping-pong pA/pB, h0 = z)
    const int initThreads = N * 16;
    const long long scatThreads = (long long)E * 16;
    const float* hin = z;
    float* bufs[2] = { pA, pB };
    for (int it = 0; it < N_ITERS; ++it) {
        float* hout = bufs[it & 1];
        prop_init_kernel<<<(initThreads + 255) / 256, 256, 0, stream>>>(hout, z, hin, dinv, N);
        prop_scatter_kernel<<<(unsigned)((scatThreads + 255) / 256), 256, 0, stream>>>(
            hout, hin, srcI, dstI, norm, E);
        hin = hout;
    }

    // 5) log-softmax -> d_out
    log_softmax_kernel<<<(N + 7) / 8, 256, 0, stream>>>(out, hin, N);
}
